// SelfAttention_21062519619631
// MI455X (gfx1250) — matrix operand order repaired and checked by reading, not compiled, not run on hardware
//
#include <hip/hip_runtime.h>
#include <hip/hip_bf16.h>
#include <math.h>

// Problem constants (match reference)
#define EMBED 1024
#define BSZ 4
#define TLEN 2048
#define MROWS (BSZ * TLEN)   // 8192 flattened (b,t) rows

typedef __attribute__((ext_vector_type(16))) _Float16 v16h;
typedef __attribute__((ext_vector_type(8)))  _Float16 v8h;
typedef __attribute__((ext_vector_type(8)))  float    v8f;

__device__ __forceinline__ v16h cat8(v8h lo, v8h hi) {
    return __builtin_shufflevector(lo, hi, 0,1,2,3,4,5,6,7,8,9,10,11,12,13,14,15);
}

// A-operand fragment: lane (llo,lhi) covers K chunks [c0+8*lhi, +7] and [c0+16+8*lhi, +7]
__device__ __forceinline__ v16h loadA(const _Float16* __restrict__ row, int c0, int lhi) {
    v8h lo = *(const v8h*)(row + c0 + 8 * lhi);
    v8h hi = *(const v8h*)(row + c0 + 16 + 8 * lhi);
    return cat8(lo, hi);
}
// B-operand fragment: lane covers K chunks [c0+8*lhi, +7] and [c0+16+8*lhi, +7]
__device__ __forceinline__ v16h loadB(const _Float16* __restrict__ row, int c0, int lhi) {
    v8h lo = *(const v8h*)(row + c0 + 8 * lhi);
    v8h hi = *(const v8h*)(row + c0 + 16 + 8 * lhi);
    return cat8(lo, hi);
}

__device__ __forceinline__ v8f wmma(v16h a, v16h b, v8f c) {
    return __builtin_amdgcn_wmma_f32_16x16x32_f16(false, a, false, b, (short)0, c, false, false);
}

// ---------------------------------------------------------------------------
// Kernel 0: f32 -> f16 convert (bandwidth-trivial pre-pass)
// ---------------------------------------------------------------------------
__global__ void cvt_f32_f16(const float* __restrict__ in, _Float16* __restrict__ out, int n) {
    int i = blockIdx.x * blockDim.x + threadIdx.x;
    if (i < n) out[i] = (_Float16)in[i];
}

// ---------------------------------------------------------------------------
// Kernel 1: projection GEMM  y[m,n] = sum_c xh[m,c] * wh[n,c] + bias[n]
// One wave -> 32x32 output macro-tile (2x2 WMMA tiles): 4 WMMA per 8 b128 loads.
// transposed==0: store f16 row-major out[m*EMBED + n]   (Q, K)
// transposed==1: store f16 as Vt[b][n][t]               (V, for att@V B-operand)
// ---------------------------------------------------------------------------
__global__ void qkv_proj(const _Float16* __restrict__ xh,
                         const _Float16* __restrict__ wh,
                         const float*    __restrict__ bias,
                         _Float16*       __restrict__ out,
                         int transposed) {
    const int lane = threadIdx.x & 31;
    const int wave = threadIdx.x >> 5;
    const int lhi  = lane >> 4;       // 0 or 1
    const int llo  = lane & 15;

    const int m0 = blockIdx.y * 32;                 // row macro-tile (over 8192)
    const int n0 = (blockIdx.x * 8 + wave) * 32;    // col macro-tile (over 1024)

    const _Float16* aRow0 = xh + (size_t)(m0 + llo) * EMBED;
    const _Float16* aRow1 = aRow0 + (size_t)16 * EMBED;
    const _Float16* bRow0 = wh + (size_t)(n0 + llo) * EMBED;
    const _Float16* bRow1 = bRow0 + (size_t)16 * EMBED;

    v8f acc00 = {}, acc01 = {}, acc10 = {}, acc11 = {};
    for (int c0 = 0; c0 < EMBED; c0 += 32) {
        __builtin_prefetch((const void*)(aRow0 + c0 + 64), 0, 3);
        v16h a0 = loadA(aRow0, c0, lhi);
        v16h a1 = loadA(aRow1, c0, lhi);
        v16h b0 = loadB(bRow0, c0, lhi);
        v16h b1 = loadB(bRow1, c0, lhi);
        acc00 = wmma(a0, b0, acc00);
        acc01 = wmma(a0, b1, acc01);
        acc10 = wmma(a1, b0, acc10);
        acc11 = wmma(a1, b1, acc11);
    }

    const float bv0 = bias[n0 + llo];
    const float bv1 = bias[n0 + 16 + llo];

    #pragma unroll
    for (int t = 0; t < 4; ++t) {
        const v8f& acc = (t == 0) ? acc00 : (t == 1) ? acc01 : (t == 2) ? acc10 : acc11;
        const int mb = m0 + ((t >> 1) << 4);
        const int nb = n0 + ((t & 1) << 4);
        const float bv = (t & 1) ? bv1 : bv0;
        #pragma unroll
        for (int r = 0; r < 8; ++r) {
            const int m = mb + r + 8 * lhi;   // D layout: VGPR r -> M=r / r+8
            const int n = nb + llo;
            const float v = acc[r] + bv;
            if (!transposed) {
                out[(size_t)m * EMBED + n] = (_Float16)v;
            } else {
                const int bb = m >> 11;          // m / TLEN
                const int tt = m & (TLEN - 1);
                out[((size_t)bb * EMBED + n) * TLEN + tt] = (_Float16)v;
            }
        }
    }
}

// ---------------------------------------------------------------------------
// Kernel 2a: S[b,q,k] = (Q[b,q,:] . K[b,k,:]) / 32, lower-triangular blocks only
// 32x32 macro-tile per wave.
// ---------------------------------------------------------------------------
__global__ void attn_scores(const _Float16* __restrict__ Qh,
                            const _Float16* __restrict__ Kh,
                            float* __restrict__ S) {
    const int lane = threadIdx.x & 31;
    const int wave = threadIdx.x >> 5;
    const int lhi  = lane >> 4;
    const int llo  = lane & 15;

    const int b  = blockIdx.z;
    const int q0 = blockIdx.y * 32;
    const int k0 = (blockIdx.x * 8 + wave) * 32;
    if (k0 > q0 + 31) return;   // macro-tile fully above the causal diagonal

    const _Float16* aRow0 = Qh + (size_t)b * TLEN * EMBED + (size_t)(q0 + llo) * EMBED;
    const _Float16* aRow1 = aRow0 + (size_t)16 * EMBED;
    const _Float16* bRow0 = Kh + (size_t)b * TLEN * EMBED + (size_t)(k0 + llo) * EMBED;
    const _Float16* bRow1 = bRow0 + (size_t)16 * EMBED;

    v8f acc00 = {}, acc01 = {}, acc10 = {}, acc11 = {};
    for (int c0 = 0; c0 < EMBED; c0 += 32) {
        v16h a0 = loadA(aRow0, c0, lhi);
        v16h a1 = loadA(aRow1, c0, lhi);
        v16h b0 = loadB(bRow0, c0, lhi);
        v16h b1 = loadB(bRow1, c0, lhi);
        acc00 = wmma(a0, b0, acc00);
        acc01 = wmma(a0, b1, acc01);
        acc10 = wmma(a1, b0, acc10);
        acc11 = wmma(a1, b1, acc11);
    }

    const float scale = 0.03125f;  // 1/sqrt(1024)
    float* sRow = S + (size_t)b * TLEN * TLEN;
    #pragma unroll
    for (int t = 0; t < 4; ++t) {
        const v8f& acc = (t == 0) ? acc00 : (t == 1) ? acc01 : (t == 2) ? acc10 : acc11;
        const int qb = q0 + ((t >> 1) << 4);
        const int kb = k0 + ((t & 1) << 4);
        #pragma unroll
        for (int r = 0; r < 8; ++r) {
            const int q = qb + r + 8 * lhi;
            const int k = kb + llo;
            float v = acc[r] * scale;
            if (k > q) v = -__builtin_inff();
            sRow[(size_t)q * TLEN + k] = v;
        }
    }
}

// ---------------------------------------------------------------------------
// Kernel 2b: per-row causal softmax; f32 scores -> f16 att (zeros past diag)
// 256 threads/row, TLEN/256 == 8 values per thread held in registers.
// ---------------------------------------------------------------------------
__global__ void row_softmax(const float* __restrict__ S, _Float16* __restrict__ att) {
    __shared__ float red[256];
    const int q = blockIdx.x;
    const int b = blockIdx.y;
    const int tid = threadIdx.x;
    const float* sRow = S + ((size_t)b * TLEN + q) * TLEN;
    _Float16* aRow = att + ((size_t)b * TLEN + q) * TLEN;

    float vals[8];
    float vmax = -__builtin_inff();
    #pragma unroll
    for (int i = 0; i < 8; ++i) {
        const int j = tid + 256 * i;
        vals[i] = (j <= q) ? sRow[j] : -__builtin_inff();
        vmax = fmaxf(vmax, vals[i]);
    }
    red[tid] = vmax; __syncthreads();
    for (int s = 128; s > 0; s >>= 1) {
        if (tid < s) red[tid] = fmaxf(red[tid], red[tid + s]);
        __syncthreads();
    }
    const float mx = red[0]; __syncthreads();

    float sum = 0.f;
    #pragma unroll
    for (int i = 0; i < 8; ++i) {
        const int j = tid + 256 * i;
        float e = (j <= q) ? __expf(vals[i] - mx) : 0.f;
        vals[i] = e;
        sum += e;
    }
    red[tid] = sum; __syncthreads();
    for (int s = 128; s > 0; s >>= 1) {
        if (tid < s) red[tid] += red[tid + s];
        __syncthreads();
    }
    const float inv = 1.f / red[0];

    #pragma unroll
    for (int i = 0; i < 8; ++i) {
        const int j = tid + 256 * i;
        aRow[j] = (_Float16)(vals[i] * inv);
    }
}

// ---------------------------------------------------------------------------
// Kernel 3: out[b,q,d] = sum_k att[b,q,k] * V[b,k,d]; B-operand from Vt[b][d][t]
// 32x32 macro-tile per wave; K loop truncated at the causal boundary.
// ---------------------------------------------------------------------------
__global__ void attn_out(const _Float16* __restrict__ att,
                         const _Float16* __restrict__ Vt,
                         float* __restrict__ out) {
    const int lane = threadIdx.x & 31;
    const int wave = threadIdx.x >> 5;
    const int lhi  = lane >> 4;
    const int llo  = lane & 15;

    const int b  = blockIdx.z;
    const int q0 = blockIdx.y * 32;
    const int n0 = (blockIdx.x * 8 + wave) * 32;   // output feature macro-tile

    const _Float16* aRow0 = att + ((size_t)b * TLEN + (q0 + llo)) * TLEN;
    const _Float16* aRow1 = aRow0 + (size_t)16 * TLEN;
    const _Float16* bRow0 = Vt + ((size_t)b * EMBED + (n0 + llo)) * TLEN;
    const _Float16* bRow1 = bRow0 + (size_t)16 * TLEN;

    const int kEnd = q0 + 32;   // multiple of 32 covering this tile's diagonal

    v8f acc00 = {}, acc01 = {}, acc10 = {}, acc11 = {};
    for (int c0 = 0; c0 < kEnd; c0 += 32) {
        v16h a0 = loadA(aRow0, c0, lhi);
        v16h a1 = loadA(aRow1, c0, lhi);
        v16h b0 = loadB(bRow0, c0, lhi);
        v16h b1 = loadB(bRow1, c0, lhi);
        acc00 = wmma(a0, b0, acc00);
        acc01 = wmma(a0, b1, acc01);
        acc10 = wmma(a1, b0, acc10);
        acc11 = wmma(a1, b1, acc11);
    }

    float* oBase = out + (size_t)b * TLEN * EMBED;
    #pragma unroll
    for (int t = 0; t < 4; ++t) {
        const v8f& acc = (t == 0) ? acc00 : (t == 1) ? acc01 : (t == 2) ? acc10 : acc11;
        const int qb = q0 + ((t >> 1) << 4);
        const int nb = n0 + ((t & 1) << 4);
        #pragma unroll
        for (int r = 0; r < 8; ++r) {
            const int q = qb + r + 8 * lhi;
            const int n = nb + llo;
            oBase[(size_t)q * EMBED + n] = acc[r];
        }
    }
}

// ---------------------------------------------------------------------------
// Host-side launch
// ---------------------------------------------------------------------------
extern "C" void kernel_launch(void* const* d_in, const int* in_sizes, int n_in,
                              void* d_out, int out_size, void* d_ws, size_t ws_size,
                              hipStream_t stream) {
    const float* x  = (const float*)d_in[0];
    const float* Wq = (const float*)d_in[1];
    const float* bq = (const float*)d_in[2];
    const float* Wk = (const float*)d_in[3];
    const float* bk = (const float*)d_in[4];
    const float* Wv = (const float*)d_in[5];
    const float* bv = (const float*)d_in[6];
    float* out = (float*)d_out;

    // Workspace layout (bytes)
    char* ws = (char*)d_ws;
    const size_t MB = 1024 * 1024;
    _Float16* xh  = (_Float16*)(ws + 0);          // 16 MB : x as f16
    _Float16* wqh = (_Float16*)(ws + 16 * MB);    //  2 MB
    _Float16* wkh = (_Float16*)(ws + 18 * MB);    //  2 MB
    _Float16* wvh = (_Float16*)(ws + 20 * MB);    //  2 MB
    _Float16* Qh  = (_Float16*)(ws + 22 * MB);    // 16 MB : Q f16 row-major
    _Float16* Kh  = (_Float16*)(ws + 38 * MB);    // 16 MB : K f16 row-major
    _Float16* Vt  = (_Float16*)(ws + 54 * MB);    // 16 MB : V f16 transposed [b][d][t]
    float*    S   = (float*)  (ws + 70 * MB);     // 64 MB : raw scores f32
    _Float16* Ath = (_Float16*)(ws + 134 * MB);   // 32 MB : softmaxed att f16

    // 0) converts
    {
        const int nx = MROWS * EMBED;             // 8388608
        cvt_f32_f16<<<nx / 256, 256, 0, stream>>>(x, xh, nx);
        const int nw = EMBED * EMBED;             // 1048576
        cvt_f32_f16<<<nw / 256, 256, 0, stream>>>(Wq, wqh, nw);
        cvt_f32_f16<<<nw / 256, 256, 0, stream>>>(Wk, wkh, nw);
        cvt_f32_f16<<<nw / 256, 256, 0, stream>>>(Wv, wvh, nw);
    }

    // 1) QKV projections: 32x32 wave-tiles -> grid = (32 n-tiles / 8 waves, 256 m-tiles)
    {
        dim3 grid(EMBED / 32 / 8, MROWS / 32);
        qkv_proj<<<grid, 256, 0, stream>>>(xh, wqh, bq, Qh, 0);
        qkv_proj<<<grid, 256, 0, stream>>>(xh, wkh, bk, Kh, 0);
        qkv_proj<<<grid, 256, 0, stream>>>(xh, wvh, bv, Vt, 1);
    }

    // 2a) scores: grid = (64 k-tiles / 8 waves, 64 q-tiles, B)
    {
        dim3 grid(TLEN / 32 / 8, TLEN / 32, BSZ);
        attn_scores<<<grid, 256, 0, stream>>>(Qh, Kh, S);
    }

    // 2b) softmax: one block per (q, b) row
    {
        dim3 grid(TLEN, BSZ);
        row_softmax<<<grid, 256, 0, stream>>>(S, Ath);
    }

    // 3) out = att @ V : grid = (32 d-tiles / 8 waves, 64 q-tiles, B)
    {
        dim3 grid(EMBED / 32 / 8, TLEN / 32, BSZ);
        attn_out<<<grid, 256, 0, stream>>>(Ath, Vt, out);
    }
}
